// LlamaDecoderLayer_5987184411368
// MI455X (gfx1250) — 16-bit and 8-bit matrix operand order repaired and checked by reading, compiled, not run on hardware
//
#include <hip/hip_runtime.h>
#include <hip/hip_bf16.h>

typedef __attribute__((ext_vector_type(16))) _Float16 v16h;
typedef __attribute__((ext_vector_type(8)))  float    v8f;
typedef __attribute__((ext_vector_type(8)))  int      v8i;

constexpr int B_ = 2, S_ = 2048, D_ = 2048, H_ = 16, HD_ = 128, FF_ = 5632;
constexpr int M_ = B_ * S_;
constexpr float EPS_ = 1e-5f;

union FragH { v16h h; int4 q[2]; };
union FragI { v8i  v; int2 d2[4]; int4 q[2]; };

#if defined(__has_builtin)
#  if __has_builtin(__builtin_amdgcn_global_load_async_to_lds_b128)
#    define HAVE_ASYNC_LDS 1
#  endif
#endif
#ifndef HAVE_ASYNC_LDS
#  define HAVE_ASYNC_LDS 0
#  warning "global_load_async_to_lds_b128 builtin unavailable: falling back to sync LDS staging"
#endif

typedef int b128v __attribute__((vector_size(16)));
typedef b128v __attribute__((address_space(1)))* gptr128;
typedef b128v __attribute__((address_space(3)))* lptr128;

__device__ __forceinline__ void copy16_to_lds(const signed char* g, signed char* l)
{
#if HAVE_ASYNC_LDS
    __builtin_amdgcn_global_load_async_to_lds_b128((gptr128)g, (lptr128)l, 0, 0);
#else
    *(int4*)l = *(const int4*)g;
#endif
}

__device__ __forceinline__ void wait_lds_copies()
{
#if HAVE_ASYNC_LDS
#  if defined(__has_builtin) && __has_builtin(__builtin_amdgcn_s_wait_asynccnt)
    __builtin_amdgcn_s_wait_asynccnt(0);
#  else
    asm volatile("s_wait_asynccnt 0" ::: "memory");
#  endif
#endif
}

__global__ __launch_bounds__(256)
void quant_weight_kernel(const float* __restrict__ w, signed char* __restrict__ q,
                         float* __restrict__ sc, int K)
{
    __shared__ float red[256];
    __shared__ float s_scale;
    const int n = blockIdx.x, tid = threadIdx.x;
    const float* wr = w + (size_t)n * K;
    float am = 0.f;
    for (int i = tid; i < K; i += 256) am = fmaxf(am, fabsf(wr[i]));
    red[tid] = am; __syncthreads();
    for (int s = 128; s > 0; s >>= 1) { if (tid < s) red[tid] = fmaxf(red[tid], red[tid+s]); __syncthreads(); }
    if (tid == 0) { float s = fmaxf(red[0], EPS_) / 127.f; s_scale = s; sc[n] = s; }
    __syncthreads();
    const float inv = 1.f / s_scale;
    signed char* qr = q + (size_t)n * K;
    for (int i = tid; i < K; i += 256) {
        int v = (int)rintf(wr[i] * inv);
        v = v < -128 ? -128 : (v > 127 ? 127 : v);
        qr[i] = (signed char)v;
    }
}

__global__ __launch_bounds__(256)
void rmsnorm_quant_kernel(const float* __restrict__ x, const float* __restrict__ g,
                          signed char* __restrict__ q, float* __restrict__ sc, int D)
{
    __shared__ float red[256];
    __shared__ float s_rstd, s_scale;
    const int t = blockIdx.x, tid = threadIdx.x;
    const float* xr = x + (size_t)t * D;
    float ss = 0.f;
    for (int i = tid; i < D; i += 256) { float v = xr[i]; ss += v * v; }
    red[tid] = ss; __syncthreads();
    for (int s = 128; s > 0; s >>= 1) { if (tid < s) red[tid] += red[tid+s]; __syncthreads(); }
    if (tid == 0) s_rstd = rsqrtf(red[0] / (float)D + EPS_);
    __syncthreads();
    const float rstd = s_rstd;
    float am = 0.f;
    for (int i = tid; i < D; i += 256) am = fmaxf(am, fabsf(xr[i] * rstd * g[i]));
    red[tid] = am; __syncthreads();
    for (int s = 128; s > 0; s >>= 1) { if (tid < s) red[tid] = fmaxf(red[tid], red[tid+s]); __syncthreads(); }
    if (tid == 0) { float s = fmaxf(red[0], EPS_) / 127.f; s_scale = s; sc[t] = s; }
    __syncthreads();
    const float inv = 1.f / s_scale;
    signed char* qr = q + (size_t)t * D;
    for (int i = tid; i < D; i += 256) {
        int v = (int)rintf(xr[i] * rstd * g[i] * inv);
        v = v < -128 ? -128 : (v > 127 ? 127 : v);
        qr[i] = (signed char)v;
    }
}

__global__ __launch_bounds__(256)
void token_quant_kernel(const float* __restrict__ x, signed char* __restrict__ q,
                        float* __restrict__ sc, int D)
{
    __shared__ float red[256];
    __shared__ float s_scale;
    const int t = blockIdx.x, tid = threadIdx.x;
    const float* xr = x + (size_t)t * D;
    float am = 0.f;
    for (int i = tid; i < D; i += 256) am = fmaxf(am, fabsf(xr[i]));
    red[tid] = am; __syncthreads();
    for (int s = 128; s > 0; s >>= 1) { if (tid < s) red[tid] = fmaxf(red[tid], red[tid+s]); __syncthreads(); }
    if (tid == 0) { float s = fmaxf(red[0], EPS_) / 127.f; s_scale = s; sc[t] = s; }
    __syncthreads();
    const float inv = 1.f / s_scale;
    signed char* qr = q + (size_t)t * D;
    for (int i = tid; i < D; i += 256) {
        int v = (int)rintf(xr[i] * inv);
        v = v < -128 ? -128 : (v > 127 ? 127 : v);
        qr[i] = (signed char)v;
    }
}

__global__ __launch_bounds__(256)
void gemm_i8_kernel(const signed char* __restrict__ Xq, const float* __restrict__ xs,
                    const signed char* __restrict__ Wq, const float* __restrict__ ws,
                    const float* __restrict__ res, float* __restrict__ out,
                    int M, int N, int K)
{
    __shared__ signed char lA[2][128 * 64];
    __shared__ signed char lB[2][64 * 64];
    const int tid = threadIdx.x;
    const int wave = tid >> 5;
    const int lane = tid & 31;
    const int col  = lane & 15;
    const int hi   = lane >> 4;
    const int mb   = blockIdx.y * 128;
    const int n0   = blockIdx.x * 64;

    const int ra0 = (tid)       >> 2, qa0 = (tid)       & 3;
    const int ra1 = (tid + 256) >> 2, qa1 = (tid + 256) & 3;
    const int rb  = tid >> 2,         qb  = tid & 3;

    auto stage = [&](int buf, int k0) {
        copy16_to_lds(&Xq[(size_t)(mb + ra0) * K + k0 + qa0 * 16], &lA[buf][ra0 * 64 + qa0 * 16]);
        copy16_to_lds(&Xq[(size_t)(mb + ra1) * K + k0 + qa1 * 16], &lA[buf][ra1 * 64 + qa1 * 16]);
        copy16_to_lds(&Wq[(size_t)(n0 + rb)  * K + k0 + qb  * 16], &lB[buf][rb  * 64 + qb  * 16]);
    };

    v8i acc[4];
#pragma unroll
    for (int n = 0; n < 4; ++n)
#pragma unroll
        for (int e = 0; e < 8; ++e) acc[n][e] = 0;

    stage(0, 0);
    int buf = 0;
    for (int k0 = 0; k0 < K; k0 += 64) {
        wait_lds_copies();
        __syncthreads();
        if (k0 + 64 < K) stage(buf ^ 1, k0 + 64);
        FragI a;
        const int ar = wave * 16 + col;
        a.d2[0] = *(const int2*)&lA[buf][ar * 64 + 0  + hi * 8];
        a.d2[1] = *(const int2*)&lA[buf][ar * 64 + 16 + hi * 8];
        a.d2[2] = *(const int2*)&lA[buf][ar * 64 + 32 + hi * 8];
        a.d2[3] = *(const int2*)&lA[buf][ar * 64 + 48 + hi * 8];
#pragma unroll
        for (int n = 0; n < 4; ++n) {
            FragI bfr;
            const int br = n * 16 + col;
            bfr.d2[0] = *(const int2*)&lB[buf][br * 64 + 0  + hi * 8];
            bfr.d2[1] = *(const int2*)&lB[buf][br * 64 + 16 + hi * 8];
            bfr.d2[2] = *(const int2*)&lB[buf][br * 64 + 32 + hi * 8];
            bfr.d2[3] = *(const int2*)&lB[buf][br * 64 + 48 + hi * 8];
            acc[n] = __builtin_amdgcn_wmma_i32_16x16x64_iu8(
                true, a.v, true, bfr.v, acc[n], false, false);
        }
        buf ^= 1;
    }
#pragma unroll
    for (int n = 0; n < 4; ++n) {
        const int nn = n0 + n * 16 + col;
        const float wsc = ws[nn];
#pragma unroll
        for (int r = 0; r < 8; ++r) {
            const int mm = mb + wave * 16 + r + hi * 8;
            float v = (float)acc[n][r] * xs[mm] * wsc;
            if (res) v += res[(size_t)mm * N + nn];
            out[(size_t)mm * N + nn] = v;
        }
    }
}

__global__ __launch_bounds__(256)
void rope_pack_kernel(const float* __restrict__ qf, const float* __restrict__ kf,
                      const float* __restrict__ vf, const int* __restrict__ pos,
                      _Float16* __restrict__ qh, _Float16* __restrict__ kh,
                      _Float16* __restrict__ vt)
{
    const size_t idx = (size_t)blockIdx.x * 256 + threadIdx.x;
    const int d = (int)(idx & (HD_ - 1));
    const int h = (int)((idx >> 7) & (H_ - 1));
    const int s = (int)((idx >> 11) & (S_ - 1));
    const int b = (int)(idx >> 22);
    const float p = (float)pos[b * S_ + s];
    const int   i = d & 63;
    const float freq = p * __expf(-(float)i * 0.14391156831212787f);
    float sn, cs;
    __sincosf(freq, &sn, &cs);
    const float qv = qf[idx], kv = kf[idx];
    const float qr = (d < 64) ? -qf[idx + 64] : qf[idx - 64];
    const float kr = (d < 64) ? -kf[idx + 64] : kf[idx - 64];
    const size_t o = (((size_t)(b * H_ + h)) * S_ + s) * HD_ + d;
    qh[o] = (_Float16)(qv * cs + qr * sn);
    kh[o] = (_Float16)(kv * cs + kr * sn);
    vt[(((size_t)(b * H_ + h)) * HD_ + d) * S_ + s] = (_Float16)vf[idx];
}

__global__ __launch_bounds__(128)
void attn_kernel(const _Float16* __restrict__ qh, const _Float16* __restrict__ kh,
                 const _Float16* __restrict__ vt, float* __restrict__ ctx)
{
    __shared__ _Float16 pl[4][16 * 32];
    const int wave = threadIdx.x >> 5;
    const int lane = threadIdx.x & 31;
    const int col  = lane & 15;
    const int hi   = lane >> 4;
    const int qbase = (blockIdx.x * 4 + wave) * 16;
    const int bh = blockIdx.y;
    const int b  = bh / H_;
    const int h  = bh % H_;
    _Float16* plw = pl[wave];
    const float SCALE = 0.0883883476483184f;

    FragH qa[4];
    {
        const char* qrow = (const char*)(qh + ((size_t)bh * S_ + qbase + col) * HD_);
#pragma unroll
        for (int c = 0; c < 4; ++c) {
            qa[c].q[0] = *(const int4*)(qrow + c * 64 + hi * 32);
            qa[c].q[1] = *(const int4*)(qrow + c * 64 + hi * 32 + 16);
        }
    }
    float mi[8], li[8];
    v8f acc[8];
#pragma unroll
    for (int r = 0; r < 8; ++r) { mi[r] = -1e30f; li[r] = 0.f; }
#pragma unroll
    for (int n = 0; n < 8; ++n)
#pragma unroll
        for (int e = 0; e < 8; ++e) acc[n][e] = 0.f;

    for (int kb0 = 0; kb0 < qbase + 16; kb0 += 32) {
        v8f st[2];
#pragma unroll
        for (int t = 0; t < 2; ++t) {
#pragma unroll
            for (int e = 0; e < 8; ++e) st[t][e] = 0.f;
            const int key = kb0 + t * 16 + col;
            const char* krow = (const char*)(kh + ((size_t)bh * S_ + key) * HD_);
#pragma unroll
            for (int c = 0; c < 4; ++c) {
                FragH kf_;
                kf_.q[0] = *(const int4*)(krow + c * 64 + hi * 32);
                kf_.q[1] = *(const int4*)(krow + c * 64 + hi * 32 + 16);
                st[t] = __builtin_amdgcn_wmma_f32_16x16x32_f16(
                    false, qa[c].h, false, kf_.h, (short)0, st[t], false, false);
            }
        }
#pragma unroll
        for (int t = 0; t < 2; ++t) {
            const int key = kb0 + t * 16 + col;
#pragma unroll
            for (int r = 0; r < 8; ++r) {
                const int qm = qbase + r + hi * 8;
                st[t][r] = (key > qm) ? -1e30f : st[t][r] * SCALE;
            }
        }
        float alpha[8];
#pragma unroll
        for (int r = 0; r < 8; ++r) {
            float v = fmaxf(st[0][r], st[1][r]);
            v = fmaxf(v, __shfl_xor(v, 1));
            v = fmaxf(v, __shfl_xor(v, 2));
            v = fmaxf(v, __shfl_xor(v, 4));
            v = fmaxf(v, __shfl_xor(v, 8));
            const float nm = fmaxf(mi[r], v);
            alpha[r] = __expf(mi[r] - nm);
            mi[r] = nm;
        }
#pragma unroll
        for (int t = 0; t < 2; ++t)
#pragma unroll
            for (int r = 0; r < 8; ++r)
                st[t][r] = __expf(st[t][r] - mi[r]);
#pragma unroll
        for (int r = 0; r < 8; ++r) {
            float v = st[0][r] + st[1][r];
            v += __shfl_xor(v, 1);
            v += __shfl_xor(v, 2);
            v += __shfl_xor(v, 4);
            v += __shfl_xor(v, 8);
            li[r] = li[r] * alpha[r] + v;
        }
#pragma unroll
        for (int n = 0; n < 8; ++n)
#pragma unroll
            for (int r = 0; r < 8; ++r)
                acc[n][r] *= alpha[r];
#pragma unroll
        for (int t = 0; t < 2; ++t)
#pragma unroll
            for (int r = 0; r < 8; ++r)
                plw[(r + hi * 8) * 32 + t * 16 + col] = (_Float16)st[t][r];
        asm volatile("s_wait_dscnt 0" ::: "memory");
        FragH pa;
        pa.q[0] = *(const int4*)&plw[col * 32 + hi * 16];
        pa.q[1] = *(const int4*)&plw[col * 32 + hi * 16 + 8];
#pragma unroll
        for (int n = 0; n < 8; ++n) {
            const char* vrow =
                (const char*)(vt + ((size_t)bh * HD_ + n * 16 + col) * S_ + kb0);
            FragH vb;
            vb.q[0] = *(const int4*)(vrow + hi * 32);
            vb.q[1] = *(const int4*)(vrow + hi * 32 + 16);
            acc[n] = __builtin_amdgcn_wmma_f32_16x16x32_f16(
                false, pa.h, false, vb.h, (short)0, acc[n], false, false);
        }
    }
#pragma unroll
    for (int r = 0; r < 8; ++r) {
        const int row = qbase + r + hi * 8;
        const float inv = 1.0f / li[r];
#pragma unroll
        for (int n = 0; n < 8; ++n)
            ctx[((size_t)(b * S_ + row)) * D_ + h * HD_ + n * 16 + col] = acc[n][r] * inv;
    }
}

__global__ __launch_bounds__(256)
void silu_mul_kernel(float* __restrict__ gate, const float* __restrict__ up)
{
    const size_t idx = (size_t)blockIdx.x * 256 + threadIdx.x;
    const float g = gate[idx];
    gate[idx] = g / (1.f + __expf(-g)) * up[idx];
}

extern "C" void kernel_launch(void* const* d_in, const int* in_sizes, int n_in,
                              void* d_out, int out_size, void* d_ws, size_t ws_size,
                              hipStream_t stream)
{
    (void)in_sizes; (void)n_in; (void)out_size; (void)ws_size;
    const float* hidden = (const float*)d_in[0];
    const int*   pos    = (const int*)d_in[1];
    const float* ln1    = (const float*)d_in[2];
    const float* ln2    = (const float*)d_in[3];
    const float* wq     = (const float*)d_in[4];
    const float* wk     = (const float*)d_in[5];
    const float* wv     = (const float*)d_in[6];
    const float* wo     = (const float*)d_in[7];
    const float* wg     = (const float*)d_in[8];
    const float* wu     = (const float*)d_in[9];
    const float* wd     = (const float*)d_in[10];
    float* out = (float*)d_out;

    char* p = (char*)d_ws;
    auto alloc = [&](size_t n) { char* r = p; p += (n + 255) & ~(size_t)255; return r; };

    signed char* qWq = (signed char*)alloc((size_t)D_ * D_);
    signed char* qWk = (signed char*)alloc((size_t)D_ * D_);
    signed char* qWv = (signed char*)alloc((size_t)D_ * D_);
    signed char* qWo = (signed char*)alloc((size_t)D_ * D_);
    signed char* qWg = (signed char*)alloc((size_t)FF_ * D_);
    signed char* qWu = (signed char*)alloc((size_t)FF_ * D_);
    signed char* qWd = (signed char*)alloc((size_t)D_ * FF_);
    float* sWq = (float*)alloc(D_ * 4);   float* sWk = (float*)alloc(D_ * 4);
    float* sWv = (float*)alloc(D_ * 4);   float* sWo = (float*)alloc(D_ * 4);
    float* sWg = (float*)alloc(FF_ * 4);  float* sWu = (float*)alloc(FF_ * 4);
    float* sWd = (float*)alloc(D_ * 4);
    signed char* xq = (signed char*)alloc((size_t)M_ * D_);
    float* xs = (float*)alloc(M_ * 4);
    float* qf = (float*)alloc((size_t)M_ * D_ * 4);
    float* kf = (float*)alloc((size_t)M_ * D_ * 4);
    float* vf = (float*)alloc((size_t)M_ * D_ * 4);
    _Float16* qhb = (_Float16*)alloc((size_t)M_ * D_ * 2);
    _Float16* khb = (_Float16*)alloc((size_t)M_ * D_ * 2);
    _Float16* vtb = (_Float16*)alloc((size_t)M_ * D_ * 2);
    float* ctx = (float*)alloc((size_t)M_ * D_ * 4);
    signed char* aq = (signed char*)alloc((size_t)M_ * D_);
    float* asc = (float*)alloc(M_ * 4);
    float* hbuf = (float*)alloc((size_t)M_ * D_ * 4);
    signed char* mq = (signed char*)alloc((size_t)M_ * D_);
    float* ms = (float*)alloc(M_ * 4);
    float* gate = (float*)alloc((size_t)M_ * FF_ * 4);
    float* up   = (float*)alloc((size_t)M_ * FF_ * 4);
    signed char* sq = (signed char*)alloc((size_t)M_ * FF_);
    float* ssc = (float*)alloc(M_ * 4);

    quant_weight_kernel<<<D_, 256, 0, stream>>>(wq, qWq, sWq, D_);
    quant_weight_kernel<<<D_, 256, 0, stream>>>(wk, qWk, sWk, D_);
    quant_weight_kernel<<<D_, 256, 0, stream>>>(wv, qWv, sWv, D_);
    quant_weight_kernel<<<D_, 256, 0, stream>>>(wo, qWo, sWo, D_);
    quant_weight_kernel<<<FF_, 256, 0, stream>>>(wg, qWg, sWg, D_);
    quant_weight_kernel<<<FF_, 256, 0, stream>>>(wu, qWu, sWu, D_);
    quant_weight_kernel<<<D_, 256, 0, stream>>>(wd, qWd, sWd, FF_);

    rmsnorm_quant_kernel<<<M_, 256, 0, stream>>>(hidden, ln1, xq, xs, D_);
    const dim3 blk(256);
    const dim3 gD(D_ / 64, M_ / 128);
    gemm_i8_kernel<<<gD, blk, 0, stream>>>(xq, xs, qWq, sWq, nullptr, qf, M_, D_, D_);
    gemm_i8_kernel<<<gD, blk, 0, stream>>>(xq, xs, qWk, sWk, nullptr, kf, M_, D_, D_);
    gemm_i8_kernel<<<gD, blk, 0, stream>>>(xq, xs, qWv, sWv, nullptr, vf, M_, D_, D_);
    rope_pack_kernel<<<(M_ * (D_ / 256)), 256, 0, stream>>>(qf, kf, vf, pos, qhb, khb, vtb);
    attn_kernel<<<dim3(S_ / 64, B_ * H_), 128, 0, stream>>>(qhb, khb, vtb, ctx);
    token_quant_kernel<<<M_, 256, 0, stream>>>(ctx, aq, asc, D_);
    gemm_i8_kernel<<<gD, blk, 0, stream>>>(aq, asc, qWo, sWo, hidden, hbuf, M_, D_, D_);

    rmsnorm_quant_kernel<<<M_, 256, 0, stream>>>(hbuf, ln2, mq, ms, D_);
    const dim3 gF(FF_ / 64, M_ / 128);
    gemm_i8_kernel<<<gF, blk, 0, stream>>>(mq, ms, qWg, sWg, nullptr, gate, M_, FF_, D_);
    gemm_i8_kernel<<<gF, blk, 0, stream>>>(mq, ms, qWu, sWu, nullptr, up, M_, FF_, D_);
    silu_mul_kernel<<<(M_ * (FF_ / 256)), 256, 0, stream>>>(gate, up);
    token_quant_kernel<<<M_, 256, 0, stream>>>(gate, sq, ssc, FF_);
    gemm_i8_kernel<<<gD, blk, 0, stream>>>(sq, ssc, qWd, sWd, hbuf, out, M_, D_, FF_);
}
